// DeepSeekBlock_43525198578338
// MI455X (gfx1250) — compile-verified, matrix operand order checked by reading, not run on hardware
//
#include <hip/hip_runtime.h>
#include <hip/hip_bf16.h>
#include <stdint.h>

// ---------------------------------------------------------------------------
// Problem constants (reference: B=4, T=2048, C=1024, H=16, Hkv=4, D=64,
// E=8 experts, Hh=512, top_k=2)
// ---------------------------------------------------------------------------
#define BDIM   4
#define TSEQ   2048
#define CDIM   1024
#define NH     16
#define NKV    4
#define DH     64
#define NE     8
#define HHID   512
#define BT     (BDIM * TSEQ)      // 8192 tokens
#define CKV    (NKV * DH)         // 256

// CDNA5 async global->LDS copy path (ASYNCcnt-tracked, ISA 08_async_tensor §4)
// Set to 0 to fall back to synchronous global->VGPR->LDS staging.
#define USE_ASYNC_COPY 1

typedef __attribute__((ext_vector_type(16))) __bf16 bf16x16;
typedef __attribute__((ext_vector_type(8)))  float  f32x8;

// Load a 16-element bf16 fragment (32 bytes) from a 16B-aligned pointer.
__device__ __forceinline__ bf16x16 frag_load(const __bf16* p) {
    union { uint4 u[2]; bf16x16 v; } t;
    t.u[0] = *reinterpret_cast<const uint4*>(p);
    t.u[1] = *reinterpret_cast<const uint4*>(p + 8);
    return t.v;
}

__device__ __forceinline__ f32x8 wmma_bf16(bf16x16 a, bf16x16 b, f32x8 c) {
    // v_wmma_f32_16x16x32_bf16  (D = A x B + C, f32 accumulate)
    return __builtin_amdgcn_wmma_f32_16x16x32_bf16(
        /*neg_a=*/false, a, /*neg_b=*/false, b,
        /*c_mod=*/(short)0, c, /*reuse_a=*/false, /*reuse_b=*/false);
}

// Per-lane async copy of 16B global -> LDS.  LDS destination operand is the
// 32-bit LDS byte address (low 32 bits of the flat pointer; ISA §10.2).
// Tracked by ASYNCcnt; completes in order.
__device__ __forceinline__ void async_b128(const __bf16* g, __bf16* l) {
#if USE_ASYNC_COPY
    const unsigned loff = (unsigned)(size_t)(void*)l;
    const unsigned long long ga = (unsigned long long)(size_t)(const void*)g;
    asm volatile("global_load_async_to_lds_b128 %0, %1, off"
                 :: "v"(loff), "v"(ga)
                 : "memory");
#else
    *reinterpret_cast<uint4*>(l) = *reinterpret_cast<const uint4*>(g);
#endif
}

#if USE_ASYNC_COPY
#define ASYNC_WAIT(n) asm volatile("s_wait_asynccnt " #n ::: "memory")
#else
#define ASYNC_WAIT(n)
#endif

// ---------------------------------------------------------------------------
// fp32 -> bf16 elementwise converter (weights / activations)
// ---------------------------------------------------------------------------
__global__ void k_cvt_bf16(const float* __restrict__ src, __bf16* __restrict__ dst, size_t n) {
    size_t i = (size_t)blockIdx.x * blockDim.x + threadIdx.x;
    if (i < n) dst[i] = (__bf16)src[i];
}

// ---------------------------------------------------------------------------
// RMSNorm over C=1024, output bf16 (x * rsqrt(mean(x^2)+eps) * w)
// ---------------------------------------------------------------------------
__global__ __launch_bounds__(256) void k_rmsnorm(const float* __restrict__ X,
                                                 const float* __restrict__ Wt,
                                                 __bf16* __restrict__ Out) {
    __shared__ float red[256];
    const int t   = blockIdx.x;
    const int tid = threadIdx.x;
    const float* xr = X + (size_t)t * CDIM;
    const float4 v = *reinterpret_cast<const float4*>(xr + tid * 4);
    red[tid] = v.x * v.x + v.y * v.y + v.z * v.z + v.w * v.w;
    __syncthreads();
    for (int st = 128; st > 0; st >>= 1) {
        if (tid < st) red[tid] += red[tid + st];
        __syncthreads();
    }
    const float scale = rsqrtf(red[0] * (1.0f / (float)CDIM) + 1e-6f);
    const float4 w = *reinterpret_cast<const float4*>(Wt + tid * 4);
    __bf16* o = Out + (size_t)t * CDIM + tid * 4;
    o[0] = (__bf16)(v.x * scale * w.x);
    o[1] = (__bf16)(v.y * scale * w.y);
    o[2] = (__bf16)(v.z * scale * w.z);
    o[3] = (__bf16)(v.w * scale * w.w);
}

// ---------------------------------------------------------------------------
// Tiled WMMA GEMM:  C[M,N] = A[M,K] @ W[N,K]^T   (A, W bf16 K-major)
// Block tile 128(M) x 64(N); K staged 64-wide, double-buffered in LDS with
// CDNA5 async global->LDS copies (ASYNCcnt) so stage s+1 streams in while
// stage s feeds the WMMAs.  256 threads = 8 waves (2 x 4); per wave 64x16
// output = 4 accumulators, 8 WMMAs per stage.
// mode 0: C = acc
// mode 1: C = acc + residual
// mode 2: C += rowScale[row*ss+si] * acc   (MoE expert-weighted accumulate)
// ---------------------------------------------------------------------------
__global__ __launch_bounds__(256) void k_gemm_bf16(
        const __bf16* __restrict__ A, const __bf16* __restrict__ W,
        float* __restrict__ Cout, int M, int N, int K,
        const float* __restrict__ residual,
        const float* __restrict__ rowScale, int scaleStride, int scaleIdx,
        int mode) {
    __shared__ __align__(32) __bf16 sA[2][128 * 64];   // 2 x 16 KB
    __shared__ __align__(32) __bf16 sB[2][64 * 64];    // 2 x  8 KB

    const int tid  = threadIdx.x;
    const int wave = tid >> 5;
    const int lane = tid & 31;
    const int wm   = wave & 1;        // 0..1  (64-row slabs)
    const int wn   = wave >> 1;       // 0..3  (16-col slabs)
    const int r    = lane & 15;
    const int kh   = lane >> 4;
    const int m0   = blockIdx.x * 128;
    const int n0   = blockIdx.y * 64;

    f32x8 acc[4];
#pragma unroll
    for (int i = 0; i < 4; ++i)
#pragma unroll
        for (int j = 0; j < 8; ++j) acc[i][j] = 0.0f;

    // Stage copy assignment: A 128x64 (64B/thread = 4 b128), B 64x64 (2 b128)
    const int arow = tid >> 1, ah = tid & 1;    // A: row, 32-elem col half
    const int brow = tid >> 2, bq = tid & 3;    // B: row, 16-elem col quarter

    auto copy_stage = [&](int s, int buf) {
        const int k0 = s * 64;
        const __bf16* ga = A + (size_t)(m0 + arow) * K + k0 + ah * 32;
        __bf16* la = &sA[buf][arow * 64 + ah * 32];
#pragma unroll
        for (int q = 0; q < 4; ++q) async_b128(ga + q * 8, la + q * 8);
        const __bf16* gb = W + (size_t)(n0 + brow) * K + k0 + bq * 16;
        __bf16* lb = &sB[buf][brow * 64 + bq * 16];
        async_b128(gb, lb);
        async_b128(gb + 8, lb + 8);
    };

    const int nstages = K >> 6;     // K/64 (K in {512,1024})
    copy_stage(0, 0);

    for (int s = 0; s < nstages; ++s) {
        const int buf = s & 1;
        if (s + 1 < nstages) {
            copy_stage(s + 1, buf ^ 1);   // prefetch next stage (async)
            ASYNC_WAIT(6);                // retire the 6 copies of stage s
        } else {
            ASYNC_WAIT(0);
        }
        __syncthreads();                  // whole tile visible to all waves

#pragma unroll
        for (int kk = 0; kk < 2; ++kk) {
            const bf16x16 bfrag = frag_load(&sB[buf][(wn * 16 + r) * 64 + kk * 32 + kh * 16]);
#pragma unroll
            for (int ms = 0; ms < 4; ++ms) {
                const bf16x16 afrag =
                    frag_load(&sA[buf][(wm * 64 + ms * 16 + r) * 64 + kk * 32 + kh * 16]);
                acc[ms] = wmma_bf16(afrag, bfrag, acc[ms]);
            }
        }
        __syncthreads();                  // buf free for stage s+2 copies
    }

    // Epilogue.  C/D 16x16 layout: VGPR rr, lanes 0-15: M=rr, N=lane;
    // lanes 16-31: M=8+rr, N=lane-16.
    const int col  = n0 + wn * 16 + (lane & 15);
    const int half = lane >> 4;
#pragma unroll
    for (int ms = 0; ms < 4; ++ms) {
#pragma unroll
        for (int rr = 0; rr < 8; ++rr) {
            const int row = m0 + wm * 64 + ms * 16 + half * 8 + rr;
            const size_t idx = (size_t)row * N + col;
            float v = acc[ms][rr];
            if (mode == 1) v += residual[idx];
            if (mode == 2)
                Cout[idx] += rowScale[(size_t)row * scaleStride + scaleIdx] * v;
            else
                Cout[idx] = v;
        }
    }
}

// ---------------------------------------------------------------------------
// RoPE for Q (f32 token-major [BT, H*D]) -> bf16 [B*H, T, D]
// ---------------------------------------------------------------------------
__global__ void k_rope_q(const float* __restrict__ QF, const float* __restrict__ F,
                         __bf16* __restrict__ Q16) {
    const size_t idx = (size_t)blockIdx.x * blockDim.x + threadIdx.x; // BT*NH*32 pairs
    const int i = idx & 31;
    const int h = (idx >> 5) & (NH - 1);
    const int t = (idx >> 9) & (TSEQ - 1);
    const int b = (int)(idx >> 20);
    const float re = QF[((size_t)(b * TSEQ + t)) * CDIM + h * DH + 2 * i];
    const float im = QF[((size_t)(b * TSEQ + t)) * CDIM + h * DH + 2 * i + 1];
    const float c = F[((size_t)t * 32 + i) * 2];
    const float s = F[((size_t)t * 32 + i) * 2 + 1];
    __bf16* o = Q16 + ((size_t)(b * NH + h) * TSEQ + t) * DH + 2 * i;
    o[0] = (__bf16)(re * c - im * s);
    o[1] = (__bf16)(re * s + im * c);
}

// RoPE for K (f32 [BT, Hkv*D]) -> bf16 [B*Hkv, T, D]
__global__ void k_rope_k(const float* __restrict__ KF, const float* __restrict__ F,
                         __bf16* __restrict__ K16) {
    const size_t idx = (size_t)blockIdx.x * blockDim.x + threadIdx.x; // BT*NKV*32 pairs
    const int i = idx & 31;
    const int h = (idx >> 5) & (NKV - 1);
    const int t = (idx >> 7) & (TSEQ - 1);
    const int b = (int)(idx >> 18);
    const float re = KF[((size_t)(b * TSEQ + t)) * CKV + h * DH + 2 * i];
    const float im = KF[((size_t)(b * TSEQ + t)) * CKV + h * DH + 2 * i + 1];
    const float c = F[((size_t)t * 32 + i) * 2];
    const float s = F[((size_t)t * 32 + i) * 2 + 1];
    __bf16* o = K16 + ((size_t)(b * NKV + h) * TSEQ + t) * DH + 2 * i;
    o[0] = (__bf16)(re * c - im * s);
    o[1] = (__bf16)(re * s + im * c);
}

// V (f32 [BT, Hkv*D]) -> transposed bf16 [B*Hkv, D, T]  (so P·V B-fragments
// are 16 contiguous elements along T — no transpose loads needed)
__global__ void k_v_transpose(const float* __restrict__ VF, __bf16* __restrict__ VT16) {
    const size_t idx = (size_t)blockIdx.x * blockDim.x + threadIdx.x; // BT*CKV
    const int c = idx & (CKV - 1);
    const int t = (idx >> 8) & (TSEQ - 1);
    const int b = (int)(idx >> 19);
    const int hkv = c >> 6, d = c & 63;
    VT16[((size_t)(b * NKV + hkv) * DH + d) * TSEQ + t] = (__bf16)VF[idx];
}

// ---------------------------------------------------------------------------
// Flash-attention (causal, GQA).  One wave per 16 query rows of one (b,h).
// S = Q·Kᵀ (4 WMMAs/32-key tile), online softmax with half-wave shfl_xor
// row reductions (C-layout: row m lives across lanes (m/8)*16..+15),
// P re-staged via LDS to A-fragment layout, O += P·V (4 WMMAs).
// ---------------------------------------------------------------------------
__global__ __launch_bounds__(32) void k_attn(const __bf16* __restrict__ Q,
                                             const __bf16* __restrict__ Kt,
                                             const __bf16* __restrict__ Vt,
                                             __bf16* __restrict__ Y) {
    __shared__ __align__(32) __bf16 sP[16 * 32];
    const int lane = threadIdx.x;
    const int qb   = blockIdx.x;              // query tile (T/16)
    const int bh   = blockIdx.y;              // b*NH + h
    const int b    = bh / NH, h = bh % NH;
    const int bkv  = b * NKV + (h / (NH / NKV));
    const int i0   = qb * 16;
    const int r    = lane & 15;
    const int kh   = lane >> 4;
    const int half = kh;

    const __bf16* qbase = Q  + ((size_t)bh * TSEQ + i0) * DH;
    const __bf16* kbase = Kt + (size_t)bkv * TSEQ * DH;
    const __bf16* vbase = Vt + (size_t)bkv * DH * TSEQ;

    bf16x16 aQ[2];
    aQ[0] = frag_load(qbase + (size_t)r * DH + kh * 16);
    aQ[1] = frag_load(qbase + (size_t)r * DH + 32 + kh * 16);

    f32x8 O[4];
    float m[8], l[8];
#pragma unroll
    for (int nn = 0; nn < 4; ++nn)
#pragma unroll
        for (int j = 0; j < 8; ++j) O[nn][j] = 0.0f;
#pragma unroll
    for (int j = 0; j < 8; ++j) { m[j] = -3.0e38f; l[j] = 0.0f; }

    const int jend = i0 + 16;                  // keys 0 .. i0+15 (causal)
    for (int j0 = 0; j0 < jend; j0 += 32) {
        // --- S = Q · Kᵀ  (16 x 32), f32 ---
        f32x8 S[2];
#pragma unroll
        for (int n = 0; n < 2; ++n) {
#pragma unroll
            for (int j = 0; j < 8; ++j) S[n][j] = 0.0f;
#pragma unroll
            for (int kk = 0; kk < 2; ++kk) {
                const bf16x16 bK =
                    frag_load(kbase + (size_t)(j0 + n * 16 + r) * DH + kk * 32 + kh * 16);
                S[n] = wmma_bf16(aQ[kk], bK, S[n]);
            }
        }
        // --- online softmax ---
#pragma unroll
        for (int rr = 0; rr < 8; ++rr) {
            const int qrow = i0 + half * 8 + rr;
            float s0 = S[0][rr] * 0.125f;        // 1/sqrt(64)
            float s1 = S[1][rr] * 0.125f;
            if (j0 + r > qrow)      s0 = -1.0e30f;
            if (j0 + 16 + r > qrow) s1 = -1.0e30f;
            float t = fmaxf(s0, s1);
#pragma unroll
            for (int off = 1; off < 16; off <<= 1) t = fmaxf(t, __shfl_xor(t, off, 32));
            const float mn    = fmaxf(m[rr], t);
            const float alpha = __expf(m[rr] - mn);
            m[rr] = mn;
            const float p0 = __expf(s0 - mn);
            const float p1 = __expf(s1 - mn);
            float ts = p0 + p1;
#pragma unroll
            for (int off = 1; off < 16; off <<= 1) ts += __shfl_xor(ts, off, 32);
            l[rr] = l[rr] * alpha + ts;
#pragma unroll
            for (int nn = 0; nn < 4; ++nn) O[nn][rr] *= alpha;
            sP[(half * 8 + rr) * 32 + r]      = (__bf16)p0;
            sP[(half * 8 + rr) * 32 + 16 + r] = (__bf16)p1;
        }
        __syncthreads();
        // --- O += P · V  (P: 16x32 A-frag; V pre-transposed [D, T]) ---
        const bf16x16 aP = frag_load(&sP[r * 32 + kh * 16]);
#pragma unroll
        for (int nn = 0; nn < 4; ++nn) {
            const bf16x16 bV =
                frag_load(vbase + (size_t)(nn * 16 + r) * TSEQ + j0 + kh * 16);
            O[nn] = wmma_bf16(aP, bV, O[nn]);
        }
        __syncthreads();
    }
    // --- normalize and store to token-major bf16 [BT, C] ---
#pragma unroll
    for (int rr = 0; rr < 8; ++rr) {
        const float inv = 1.0f / l[rr];
        const int trow  = i0 + half * 8 + rr;
#pragma unroll
        for (int nn = 0; nn < 4; ++nn) {
            Y[((size_t)(b * TSEQ + trow)) * CDIM + h * DH + nn * 16 + r] =
                (__bf16)(O[nn][rr] * inv);
        }
    }
}

// ---------------------------------------------------------------------------
// Router: logits = xn2 @ gate_wᵀ, top-2 softmax -> dense per-token weights
// ---------------------------------------------------------------------------
__global__ __launch_bounds__(32) void k_router(const __bf16* __restrict__ Xn2,
                                               const float* __restrict__ G,
                                               float* __restrict__ logitsOut,
                                               float* __restrict__ wfull) {
    const int t = blockIdx.x, lane = threadIdx.x;
    float acc[NE];
#pragma unroll
    for (int e = 0; e < NE; ++e) acc[e] = 0.0f;
    const __bf16* xr = Xn2 + (size_t)t * CDIM + lane * 32;
    for (int j = 0; j < 32; ++j) {
        const float x = (float)xr[j];
        const int c = lane * 32 + j;
#pragma unroll
        for (int e = 0; e < NE; ++e) acc[e] += x * G[e * CDIM + c];
    }
#pragma unroll
    for (int e = 0; e < NE; ++e)
#pragma unroll
        for (int off = 1; off < 32; off <<= 1) acc[e] += __shfl_xor(acc[e], off, 32);
    if (lane == 0) {
        int e0 = 0; float l0 = acc[0];
        for (int e = 1; e < NE; ++e) if (acc[e] > l0) { l0 = acc[e]; e0 = e; }
        int e1 = -1; float l1 = -3.0e38f;
        for (int e = 0; e < NE; ++e) if (e != e0 && acc[e] > l1) { l1 = acc[e]; e1 = e; }
        const float w0 = 1.0f / (1.0f + __expf(l1 - l0));
        const float w1 = 1.0f - w0;
        for (int e = 0; e < NE; ++e) {
            logitsOut[(size_t)t * NE + e] = acc[e];
            wfull[(size_t)t * NE + e] = (e == e0) ? w0 : ((e == e1) ? w1 : 0.0f);
        }
    }
}

// silu(g) * u -> bf16
__global__ void k_silu_mul(const float* __restrict__ g, const float* __restrict__ u,
                           __bf16* __restrict__ o, size_t n) {
    const size_t i = (size_t)blockIdx.x * blockDim.x + threadIdx.x;
    if (i < n) {
        const float gv = g[i];
        o[i] = (__bf16)((gv / (1.0f + __expf(-gv))) * u[i]);
    }
}

// out = a + b  (final residual: x_mid + (shared + experts))
__global__ void k_add2(const float* __restrict__ a, const float* __restrict__ b,
                       float* __restrict__ o, size_t n) {
    const size_t i = (size_t)blockIdx.x * blockDim.x + threadIdx.x;
    if (i < n) o[i] = a[i] + b[i];
}

// ---------------------------------------------------------------------------
// Host-side orchestration
// ---------------------------------------------------------------------------
extern "C" void kernel_launch(void* const* d_in, const int* in_sizes, int n_in,
                              void* d_out, int out_size, void* d_ws, size_t ws_size,
                              hipStream_t stream) {
    (void)in_sizes; (void)n_in; (void)out_size; (void)ws_size;

    const float* x      = (const float*)d_in[0];
    const float* freqs  = (const float*)d_in[1];
    const float* n1w    = (const float*)d_in[2];
    const float* wq     = (const float*)d_in[3];
    const float* wk     = (const float*)d_in[4];
    const float* wv     = (const float*)d_in[5];
    const float* wo     = (const float*)d_in[6];
    const float* n2w    = (const float*)d_in[7];
    const float* gate   = (const float*)d_in[8];
    const float* sw1    = (const float*)d_in[9];
    const float* sw2    = (const float*)d_in[10];
    const float* sw3    = (const float*)d_in[11];
    const float* ew1    = (const float*)d_in[12];
    const float* ew2    = (const float*)d_in[13];
    const float* ew3    = (const float*)d_in[14];

    float* out_x      = (float*)d_out;
    float* out_logits = out_x + (size_t)BT * CDIM;

    // ---- workspace layout ----
    char* base = (char*)d_ws;
    size_t off = 0;
    auto alloc = [&](size_t bytes) { size_t o = off; off += (bytes + 255) & ~(size_t)255; return o; };

    const size_t oWQ  = alloc((size_t)CDIM * CDIM * 2);
    const size_t oWK  = alloc((size_t)CKV  * CDIM * 2);
    const size_t oWV  = alloc((size_t)CKV  * CDIM * 2);
    const size_t oWO  = alloc((size_t)CDIM * CDIM * 2);
    const size_t oS1  = alloc((size_t)HHID * CDIM * 2);
    const size_t oS2  = alloc((size_t)CDIM * HHID * 2);
    const size_t oS3  = alloc((size_t)HHID * CDIM * 2);
    const size_t oE1  = alloc((size_t)NE * HHID * CDIM * 2);
    const size_t oE2  = alloc((size_t)NE * CDIM * HHID * 2);
    const size_t oE3  = alloc((size_t)NE * HHID * CDIM * 2);
    const size_t oXN  = alloc((size_t)BT * CDIM * 2);       // rmsnorm1(x) bf16
    const size_t oQF  = alloc((size_t)BT * CDIM * 4);       // q f32
    const size_t oKF  = alloc((size_t)BT * CKV  * 4);       // k f32
    const size_t oVF  = alloc((size_t)BT * CKV  * 4);       // v f32
    const size_t oQ16 = alloc((size_t)BT * CDIM * 2);       // q bf16 [B*H,T,D]
    const size_t oK16 = alloc((size_t)BT * CKV  * 2);       // k bf16 [B*Hkv,T,D]
    const size_t oVT  = alloc((size_t)BT * CKV  * 2);       // v bf16 [B*Hkv,D,T]
    const size_t oY16 = alloc((size_t)BT * CDIM * 2);       // attn out bf16
    const size_t oXM  = alloc((size_t)BT * CDIM * 4);       // x + attn  f32
    const size_t oXN2 = alloc((size_t)BT * CDIM * 2);       // rmsnorm2 bf16
    const size_t oGF  = alloc((size_t)BT * HHID * 4);       // gate proj f32
    const size_t oUF  = alloc((size_t)BT * HHID * 4);       // up proj f32
    const size_t oH16 = alloc((size_t)BT * HHID * 2);       // silu*up bf16
    const size_t oACC = alloc((size_t)BT * CDIM * 4);       // shared + experts f32
    const size_t oWF  = alloc((size_t)BT * NE * 4);         // w_full f32

    auto bf = [&](size_t o) { return (__bf16*)(base + o); };
    auto fp = [&](size_t o) { return (float*)(base + o); };

    // ---- 1) convert weights fp32 -> bf16 ----
    auto cvt = [&](const float* src, size_t n, size_t dstOff) {
        k_cvt_bf16<<<(unsigned)((n + 255) / 256), 256, 0, stream>>>(src, bf(dstOff), n);
    };
    cvt(wq,  (size_t)CDIM * CDIM, oWQ);
    cvt(wk,  (size_t)CKV  * CDIM, oWK);
    cvt(wv,  (size_t)CKV  * CDIM, oWV);
    cvt(wo,  (size_t)CDIM * CDIM, oWO);
    cvt(sw1, (size_t)HHID * CDIM, oS1);
    cvt(sw2, (size_t)CDIM * HHID, oS2);
    cvt(sw3, (size_t)HHID * CDIM, oS3);
    cvt(ew1, (size_t)NE * HHID * CDIM, oE1);
    cvt(ew2, (size_t)NE * CDIM * HHID, oE2);
    cvt(ew3, (size_t)NE * HHID * CDIM, oE3);

    auto gemm = [&](size_t aOff, size_t wOff, float* c, int M, int N, int K,
                    const float* resid, const float* rs, int ss, int si, int mode) {
        dim3 grid(M / 128, N / 64);
        k_gemm_bf16<<<grid, 256, 0, stream>>>(bf(aOff), bf(wOff), c, M, N, K,
                                              resid, rs, ss, si, mode);
    };

    // ---- 2) rmsnorm1 -> xn (bf16) ----
    k_rmsnorm<<<BT, 256, 0, stream>>>(x, n1w, bf(oXN));

    // ---- 3) QKV projections (WMMA) ----
    gemm(oXN, oWQ, fp(oQF), BT, CDIM, CDIM, nullptr, nullptr, 0, 0, 0);
    gemm(oXN, oWK, fp(oKF), BT, CKV,  CDIM, nullptr, nullptr, 0, 0, 0);
    gemm(oXN, oWV, fp(oVF), BT, CKV,  CDIM, nullptr, nullptr, 0, 0, 0);

    // ---- 4) RoPE (f32) + layout for attention ----
    k_rope_q<<<(BT * NH * 32) / 256, 256, 0, stream>>>(fp(oQF), freqs, bf(oQ16));
    k_rope_k<<<(BT * NKV * 32) / 256, 256, 0, stream>>>(fp(oKF), freqs, bf(oK16));
    k_v_transpose<<<(BT * CKV) / 256, 256, 0, stream>>>(fp(oVF), bf(oVT));

    // ---- 5) causal flash attention (WMMA) ----
    {
        dim3 grid(TSEQ / 16, BDIM * NH);
        k_attn<<<grid, 32, 0, stream>>>(bf(oQ16), bf(oK16), bf(oVT), bf(oY16));
    }

    // ---- 6) out-proj + residual: x_mid = y @ wo^T + x ----
    gemm(oY16, oWO, fp(oXM), BT, CDIM, CDIM, x, nullptr, 0, 0, 1);

    // ---- 7) rmsnorm2 -> xn2 (bf16) ----
    k_rmsnorm<<<BT, 256, 0, stream>>>(fp(oXM), n2w, bf(oXN2));

    // ---- 8) router (logits straight to d_out) + top-2 weights ----
    k_router<<<BT, 32, 0, stream>>>(bf(oXN2), gate, out_logits, fp(oWF));

    // ---- 9) shared SwiGLU: ACC = (silu(xn2@sw1ᵀ) * (xn2@sw3ᵀ)) @ sw2ᵀ ----
    gemm(oXN2, oS1, fp(oGF), BT, HHID, CDIM, nullptr, nullptr, 0, 0, 0);
    gemm(oXN2, oS3, fp(oUF), BT, HHID, CDIM, nullptr, nullptr, 0, 0, 0);
    k_silu_mul<<<(BT * HHID) / 256, 256, 0, stream>>>(fp(oGF), fp(oUF), bf(oH16),
                                                      (size_t)BT * HHID);
    gemm(oH16, oS2, fp(oACC), BT, CDIM, HHID, nullptr, nullptr, 0, 0, 0); // init ACC

    // ---- 10) MoE: dense over all experts (reference-exact), weighted
    //           accumulate into ACC via per-row w_full[:, e] ----
    for (int e = 0; e < NE; ++e) {
        const size_t e1o = oE1 + (size_t)e * HHID * CDIM * 2;
        const size_t e3o = oE3 + (size_t)e * HHID * CDIM * 2;
        const size_t e2o = oE2 + (size_t)e * CDIM * HHID * 2;
        gemm(oXN2, e1o, fp(oGF), BT, HHID, CDIM, nullptr, nullptr, 0, 0, 0);
        gemm(oXN2, e3o, fp(oUF), BT, HHID, CDIM, nullptr, nullptr, 0, 0, 0);
        k_silu_mul<<<(BT * HHID) / 256, 256, 0, stream>>>(fp(oGF), fp(oUF), bf(oH16),
                                                          (size_t)BT * HHID);
        gemm(oH16, e2o, fp(oACC), BT, CDIM, HHID, nullptr, fp(oWF), NE, e, 2);
    }

    // ---- 11) final residual: out = x_mid + ACC ----
    k_add2<<<(BT * CDIM) / 256, 256, 0, stream>>>(fp(oXM), fp(oACC), out_x,
                                                  (size_t)BT * CDIM);
}
